// LSTMModel_86500641341559
// MI455X (gfx1250) — matrix operand order repaired and checked by reading, not compiled, not run on hardware
//
#include <hip/hip_runtime.h>
#include <hip/hip_bf16.h>

// ---------------------------------------------------------------------------
// Types for CDNA5 WMMA (wave32): A/B = v16h (16 x f16), C/D = v8f (8 x f32)
// ---------------------------------------------------------------------------
typedef __attribute__((ext_vector_type(16))) _Float16 v16h;
typedef __attribute__((ext_vector_type(8)))  _Float16 v8h;
typedef __attribute__((ext_vector_type(8)))  float    v8f;

// Problem constants
constexpr int Bb = 256;   // batch
constexpr int Tt = 512;   // time
constexpr int Ii = 64;    // input
constexpr int Hh = 128;   // hidden
constexpr int Gg = 512;   // 4*H gate width
constexpr int H2 = 256;   // 2*H

static __device__ __forceinline__ v16h combine16(v8h lo, v8h hi) {
    v16h r;
#pragma unroll
    for (int i = 0; i < 8; ++i) { r[i] = lo[i]; r[i + 8] = hi[i]; }
    return r;
}

static __device__ __forceinline__ float sigm(float x) {
    return 1.0f / (1.0f + __expf(-x));
}

// ---------------------------------------------------------------------------
// f32 -> f16 conversion
// ---------------------------------------------------------------------------
__global__ void f32_to_f16_kernel(const float* __restrict__ src,
                                  _Float16* __restrict__ dst, long n) {
    long i = (long)blockIdx.x * blockDim.x + threadIdx.x;
    if (i < n) dst[i] = (_Float16)src[i];
}

// ---------------------------------------------------------------------------
// Generic WMMA GEMM:  out[m][n] = sum_k A16[m][k] * W16[n][k] + b1[n] + b2[n]
//   A16: f16, row m at A + m*astride, K contiguous
//   W16: f16 [512][K] row-major  (acts as Bᵀ: B[k][n] = W[n][k])
//   out: f16 [M][512]
// One wave computes one 16x16 tile (K looped in steps of 32).
// Fragment layouts used here (contiguous K, same order in A and B):
//   A 16x32 f16: lane holds row m=lane&15, 16 consecutive K (0..15 / 16..31)
//   B 32x16 f16: lane holds column n=lane&15, 16 consecutive K (0..15 / 16..31)
//   C/D f32:     lane holds column n=lane&15, 8 rows (v or v+8 by lane half)
// ---------------------------------------------------------------------------
__global__ __launch_bounds__(256) void gemm_xg_kernel(
    const _Float16* __restrict__ A, long astride, int M, int K,
    const _Float16* __restrict__ W,
    const float* __restrict__ b1, const float* __restrict__ b2,
    _Float16* __restrict__ out) {
    const int lane = threadIdx.x & 31;
    const int wave = threadIdx.x >> 5;
    const long wt  = (long)blockIdx.x * 8 + wave;
    const long tm  = wt / (Gg / 16);
    const int  tn  = (int)(wt % (Gg / 16));
    if (tm * 16 >= (long)M) return;

    const int l15 = lane & 15;
    const int hiL = lane >> 4;            // 0: lanes 0-15, 1: lanes 16-31
    const int g0  = hiL ? 8 : 0;          // unused; A fragment also uses kb
    const int kb  = hiL ? 16 : 0;         // B fragment K sub-offset
    const int arow = (int)(tm * 16) + l15;
    const int n    = tn * 16 + l15;

    v8f acc = {};
    for (int k = 0; k < K; k += 32) {
        const _Float16* pa = A + (long)arow * astride + k + kb;
        v16h af = *(const v16h*)pa;
        v16h bf = *(const v16h*)(W + (long)n * K + k + kb);
        acc = __builtin_amdgcn_wmma_f32_16x16x32_f16(
            false, af, false, bf, (short)0, acc, false, false);
    }

    const float bias = b1[n] + b2[n];
#pragma unroll
    for (int v = 0; v < 8; ++v) {
        const int mr = (int)(tm * 16) + (hiL ? v + 8 : v);
        out[(long)mr * Gg + n] = (_Float16)(acc[v] + bias);
    }
}

// ---------------------------------------------------------------------------
// Persistent bidirectional LSTM direction kernel.
//   grid.x = #directions (block 0 / block 1). 512 threads = 16 waves.
//   xg:  f16 [B][512(T)][512(G)]  -- x-projection incl. both biases
//   Whh: f16 [512][128]
//   hout: f16 [B][T][2H]   (written when store_all)
//   hlast: f32 [B][2H]     (written when !store_all, final h)
// LDS (dynamic, 256 KB): h double buffer 2x(256*128 f16) + gate chunk 64x512 f32
// c state lives in registers (creg[4][16] per thread) across the whole scan.
// Per step: 128 v_wmma ops (M=256,N=512,K=128), barrier, VALU gate math.
// ---------------------------------------------------------------------------
__global__ __launch_bounds__(512) void lstm_dir_kernel(
    const _Float16* __restrict__ xg0, const _Float16* __restrict__ xg1,
    const _Float16* __restrict__ Whh0, const _Float16* __restrict__ Whh1,
    _Float16* __restrict__ hout, float* __restrict__ hlast,
    int Tloop, int rev0, int rev1, int store_all) {
    extern __shared__ unsigned char smem[];
    _Float16* hbuf  = (_Float16*)smem;                    // 2 * 256*128 f16
    float*    gates = (float*)(smem + 2 * Bb * Hh * 2);   // 64*512 f32

    const int dir = blockIdx.x;
    const _Float16* xg  = dir ? xg1 : xg0;
    const _Float16* Whh = dir ? Whh1 : Whh0;
    const int reverse   = dir ? rev1 : rev0;

    const int tid  = threadIdx.x;
    const int lane = tid & 31;
    const int wave = tid >> 5;
    const int l15  = lane & 15;
    const int hiL  = lane >> 4;
    const int g0   = hiL ? 8 : 0;
    const int kb   = hiL ? 16 : 0;

    // zero-init h buffer 0 and cell state
    for (int i = tid; i < Bb * Hh; i += 512) hbuf[i] = (_Float16)0.0f;
    float creg[4][16];
#pragma unroll
    for (int c0 = 0; c0 < 4; ++c0)
#pragma unroll
        for (int q = 0; q < 16; ++q) creg[c0][q] = 0.0f;
    __syncthreads();

    int cur = 0;
    for (int s = 0; s < Tloop; ++s) {
        const int t = reverse ? (Tt - 1 - s) : s;
        if (s + 1 < Tloop) {   // prefetch next step's xg slice (global_prefetch_b8)
            const int tn2 = reverse ? (Tt - 2 - s) : (s + 1);
            const _Float16* pf = xg + (size_t)(tid & 255) * (Tt * Gg) +
                                 (size_t)tn2 * Gg + ((tid >> 8) << 8);
            __builtin_prefetch(pf, 0, 1);
        }
        _Float16* hprev = hbuf + cur * (Bb * Hh);
        _Float16* hnew  = hbuf + (cur ^ 1) * (Bb * Hh);

#pragma unroll
        for (int chunk = 0; chunk < 4; ++chunk) {
            // ---- GEMM phase: gates[0:64][0:512] = h_prev @ Whhᵀ + xg_t ----
            // wave w covers N tiles {2w, 2w+1} for all 4 M tiles of the chunk
#pragma unroll
            for (int j = 0; j < 2; ++j) {
                const int tn = wave * 2 + j;
                const int n  = tn * 16 + l15;
#pragma unroll
                for (int mt = 0; mt < 4; ++mt) {
                    const int rl = mt * 16 + l15;        // A row within chunk
                    const int ra = chunk * 64 + rl;      // absolute batch row
                    v8f acc = {};
#pragma unroll
                    for (int k = 0; k < Hh; k += 32) {
                        const _Float16* pa = hprev + ra * Hh + k + kb;
                        v16h af = *(const v16h*)pa;
                        v16h bf = *(const v16h*)(Whh + (size_t)n * Hh + k + kb);
                        acc = __builtin_amdgcn_wmma_f32_16x16x32_f16(
                            false, af, false, bf, (short)0, acc, false, false);
                    }
#pragma unroll
                    for (int v = 0; v < 8; ++v) {
                        const int rc = mt * 16 + (hiL ? v + 8 : v);
                        const int b  = chunk * 64 + rc;
                        const float xv =
                            (float)xg[(size_t)b * (Tt * Gg) + (size_t)t * Gg + n];
                        gates[rc * Gg + n] = acc[v] + xv;
                    }
                }
            }
            __syncthreads();

            // ---- elementwise gate phase: c in registers, h -> LDS (+global) ----
#pragma unroll
            for (int q = 0; q < 16; ++q) {
                const int idx = q * 512 + tid;           // 64*128 elements
                const int rc  = idx >> 7;
                const int col = idx & 127;
                const int b   = chunk * 64 + rc;
                const float gi = gates[rc * Gg + col];
                const float gf = gates[rc * Gg + 128 + col];
                const float gg = gates[rc * Gg + 256 + col];
                const float go = gates[rc * Gg + 384 + col];
                float c = sigm(gf) * creg[chunk][q] + sigm(gi) * tanhf(gg);
                creg[chunk][q] = c;
                const float h = sigm(go) * tanhf(c);
                hnew[b * Hh + col] = (_Float16)h;
                if (store_all)
                    hout[((size_t)b * Tt + t) * H2 + dir * Hh + col] = (_Float16)h;
            }
            __syncthreads();
        }
        cur ^= 1;
    }

    if (!store_all) {
        const _Float16* hf = hbuf + cur * (Bb * Hh);
        for (int i = tid; i < Bb * Hh; i += 512) {
            const int b = i >> 7, col = i & 127;
            hlast[b * H2 + dir * Hh + col] = (float)hf[i];
        }
    }
}

// ---------------------------------------------------------------------------
// Layer-1 backward contributes only its t=T-1 hidden state (one step from
// zero state): c = sig(i)*tanh(g), h = sig(o)*tanh(c).
// ---------------------------------------------------------------------------
__global__ void l1rev_last_kernel(const _Float16* __restrict__ xg /*[256][512]*/,
                                  float* __restrict__ fcin /*[256][256]*/) {
    const int idx = blockIdx.x * blockDim.x + threadIdx.x;
    if (idx >= Bb * Hh) return;
    const int b = idx >> 7, j = idx & 127;
    const float gi = (float)xg[b * Gg + j];
    const float gg = (float)xg[b * Gg + 256 + j];
    const float go = (float)xg[b * Gg + 384 + j];
    const float c  = sigm(gi) * tanhf(gg);
    fcin[b * H2 + Hh + j] = sigm(go) * tanhf(c);
}

// ---------------------------------------------------------------------------
// FC head: out[b] = fc2_b + sum_j relu(fc_in[b]·fc1_w[j] + fc1_b[j]) * fc2_w[j]
// ---------------------------------------------------------------------------
__global__ void fc_kernel(const float* __restrict__ fcin,
                          const float* __restrict__ w1, const float* __restrict__ b1,
                          const float* __restrict__ w2, const float* __restrict__ b2,
                          float* __restrict__ out) {
    const int b = blockIdx.x * blockDim.x + threadIdx.x;
    if (b >= Bb) return;
    float acc2 = b2[0];
    for (int j = 0; j < Hh; ++j) {
        float a = b1[j];
        for (int k = 0; k < H2; ++k) a = fmaf(fcin[b * H2 + k], w1[j * H2 + k], a);
        acc2 = fmaf(fmaxf(a, 0.0f), w2[j], acc2);
    }
    out[b] = acc2;
}

// ---------------------------------------------------------------------------
// Host orchestration
// ---------------------------------------------------------------------------
extern "C" void kernel_launch(void* const* d_in, const int* in_sizes, int n_in,
                              void* d_out, int out_size, void* d_ws, size_t ws_size,
                              hipStream_t stream) {
    (void)in_sizes; (void)n_in; (void)out_size; (void)ws_size;
    const float* x        = (const float*)d_in[0];
    const float* Wih_l0   = (const float*)d_in[1];
    const float* Whh_l0   = (const float*)d_in[2];
    const float* bih_l0   = (const float*)d_in[3];
    const float* bhh_l0   = (const float*)d_in[4];
    const float* Wih_l0r  = (const float*)d_in[5];
    const float* Whh_l0r  = (const float*)d_in[6];
    const float* bih_l0r  = (const float*)d_in[7];
    const float* bhh_l0r  = (const float*)d_in[8];
    const float* Wih_l1   = (const float*)d_in[9];
    const float* Whh_l1   = (const float*)d_in[10];
    const float* bih_l1   = (const float*)d_in[11];
    const float* bhh_l1   = (const float*)d_in[12];
    const float* Wih_l1r  = (const float*)d_in[13];
    // d_in[14] = W_hh_l1r unused: layer-1 backward is one step from h=0
    const float* bih_l1r  = (const float*)d_in[15];
    const float* bhh_l1r  = (const float*)d_in[16];
    const float* fc1_w    = (const float*)d_in[17];
    const float* fc1_b    = (const float*)d_in[18];
    const float* fc2_w    = (const float*)d_in[19];
    const float* fc2_b    = (const float*)d_in[20];

    // ---- workspace layout ----
    char* ws = (char*)d_ws;
    size_t off = 0;
    auto take = [&](size_t bytes) -> char* {
        char* p = ws + off;
        off = (off + bytes + 255) & ~(size_t)255;
        return p;
    };
    _Float16* x16    = (_Float16*)take((size_t)Bb * Tt * Ii * 2);   // 16.8 MB
    _Float16* wih0   = (_Float16*)take((size_t)Gg * Ii * 2);
    _Float16* wih0r  = (_Float16*)take((size_t)Gg * Ii * 2);
    _Float16* whh0   = (_Float16*)take((size_t)Gg * Hh * 2);
    _Float16* whh0r  = (_Float16*)take((size_t)Gg * Hh * 2);
    _Float16* wih1   = (_Float16*)take((size_t)Gg * H2 * 2);
    _Float16* wih1r  = (_Float16*)take((size_t)Gg * H2 * 2);
    _Float16* whh1   = (_Float16*)take((size_t)Gg * Hh * 2);
    _Float16* xg0f   = (_Float16*)take((size_t)Bb * Tt * Gg * 2);   // 134 MB
    _Float16* xg0r   = (_Float16*)take((size_t)Bb * Tt * Gg * 2);   // 134 MB
    _Float16* h0     = (_Float16*)take((size_t)Bb * Tt * H2 * 2);   // 67 MB
    float*    fcin   = (float*)take((size_t)Bb * H2 * 4);
    _Float16* xg1f   = xg0f;            // reuse: xg0 buffers dead after layer 0
    _Float16* xg1r   = xg0r;            // only [256][512] used

    // ---- f32 -> f16 conversions ----
    auto cvt = [&](const float* s, _Float16* d, long n) {
        f32_to_f16_kernel<<<(unsigned)((n + 255) / 256), 256, 0, stream>>>(s, d, n);
    };
    cvt(x,       x16,   (long)Bb * Tt * Ii);
    cvt(Wih_l0,  wih0,  (long)Gg * Ii);
    cvt(Wih_l0r, wih0r, (long)Gg * Ii);
    cvt(Whh_l0,  whh0,  (long)Gg * Hh);
    cvt(Whh_l0r, whh0r, (long)Gg * Hh);
    cvt(Wih_l1,  wih1,  (long)Gg * H2);
    cvt(Wih_l1r, wih1r, (long)Gg * H2);
    cvt(Whh_l1,  whh1,  (long)Gg * Hh);

    const int M   = Bb * Tt;                          // 131072
    const unsigned gemmBlocksBig = (unsigned)(((long)(M / 16) * (Gg / 16) + 7) / 8);
    const unsigned gemmBlocksSm  = (unsigned)(((Bb / 16) * (Gg / 16) + 7) / 8);

    // ---- layer 0 input projections (both biases folded in) ----
    gemm_xg_kernel<<<gemmBlocksBig, 256, 0, stream>>>(
        x16, (long)Ii, M, Ii, wih0, bih_l0, bhh_l0, xg0f);
    gemm_xg_kernel<<<gemmBlocksBig, 256, 0, stream>>>(
        x16, (long)Ii, M, Ii, wih0r, bih_l0r, bhh_l0r, xg0r);

    // ---- layer 0 recurrence: 2 blocks = fwd + rev, persistent ----
    (void)hipFuncSetAttribute(reinterpret_cast<const void*>(lstm_dir_kernel),
                              hipFuncAttributeMaxDynamicSharedMemorySize, 262144);
    lstm_dir_kernel<<<2, 512, 262144, stream>>>(
        xg0f, xg0r, whh0, whh0r, h0, fcin, Tt, /*rev0=*/0, /*rev1=*/1,
        /*store_all=*/1);

    // ---- layer 1 input projections ----
    gemm_xg_kernel<<<gemmBlocksBig, 256, 0, stream>>>(
        h0, (long)H2, M, H2, wih1, bih_l1, bhh_l1, xg1f);
    // backward dir: only row (b, T-1) needed -> M=256, stride T*2H
    gemm_xg_kernel<<<gemmBlocksSm, 256, 0, stream>>>(
        h0 + (size_t)(Tt - 1) * H2, (long)Tt * H2, Bb, H2, wih1r,
        bih_l1r, bhh_l1r, xg1r);

    // ---- layer 1 forward recurrence (final h -> fcin[:, 0:128]) ----
    lstm_dir_kernel<<<1, 512, 262144, stream>>>(
        xg1f, xg1f, whh1, whh1, h0 /*unused*/, fcin, Tt, 0, 0, /*store_all=*/0);

    // ---- layer 1 backward single step (-> fcin[:, 128:256]) ----
    l1rev_last_kernel<<<(Bb * Hh + 255) / 256, 256, 0, stream>>>(xg1r, fcin);

    // ---- FC head ----
    fc_kernel<<<1, 256, 0, stream>>>(fcin, fc1_w, fc1_b, fc2_w, fc2_b,
                                     (float*)d_out);
}
